// GCN_JK_74698071212049
// MI455X (gfx1250) — compile-verified, matrix operand order checked by reading, not run on hardware
//
#include <hip/hip_runtime.h>
#include <hip/hip_bf16.h>

typedef __attribute__((ext_vector_type(16))) __bf16 v16bf;
typedef __attribute__((ext_vector_type(8)))  __bf16 v8bf;
typedef __attribute__((ext_vector_type(8)))  float  v8f;

// ---------- WMMA fragment loaders (CDNA5 ISA 7.12.2 16-bit layouts) ----------
// Per lane-half, a 16-bit A/B fragment covers K = {ka..ka+7} and {ka+16..ka+23}
// (ka = k0 + 8*half), i.e. two contiguous runs of 8.
__device__ __forceinline__ v16bf load_a_f32(const float* __restrict__ p) {
  float4 a0 = *(const float4*)(p);
  float4 a1 = *(const float4*)(p + 4);
  float4 a2 = *(const float4*)(p + 16);
  float4 a3 = *(const float4*)(p + 20);
  v16bf v = {(__bf16)a0.x, (__bf16)a0.y, (__bf16)a0.z, (__bf16)a0.w,
             (__bf16)a1.x, (__bf16)a1.y, (__bf16)a1.z, (__bf16)a1.w,
             (__bf16)a2.x, (__bf16)a2.y, (__bf16)a2.z, (__bf16)a2.w,
             (__bf16)a3.x, (__bf16)a3.y, (__bf16)a3.z, (__bf16)a3.w};
  return v;
}
__device__ __forceinline__ v16bf load_b_bf16(const __bf16* __restrict__ p) {
  union { v16bf v; v8bf h[2]; } u;
  u.h[0] = *(const v8bf*)(p);        // 16-byte load
  u.h[1] = *(const v8bf*)(p + 16);   // 16-byte load
  return u.v;
}

// ---------- one-shot weight prep: W[K x H] f32 -> Wt[Hpad x K] bf16 (zero pad) ----------
__global__ void k_wt_bf16(const float* __restrict__ W, __bf16* __restrict__ Wt,
                          int K, int H, int Hpad) {
  int i = blockIdx.x * blockDim.x + threadIdx.x;
  if (i >= Hpad * K) return;
  int h = i / K, k = i - h * K;
  Wt[i] = (__bf16)((h < H) ? W[k * H + h] : 0.0f);
}
__global__ void k_pad_bias(const float* __restrict__ b, float* __restrict__ bp,
                           int C, int Cpad) {
  int i = threadIdx.x;
  if (i < Cpad) bp[i] = (i < C) ? b[i] : 0.0f;
}

// ---------- degree / normalization ----------
__global__ void k_init_deg(float* __restrict__ deg, int n) {
  int i = blockIdx.x * blockDim.x + threadIdx.x;
  if (i < n) deg[i] = 1.0f;                   // self loop
}
__global__ void k_deg_scatter(const int* __restrict__ dst, float* __restrict__ deg, int e) {
  int i = blockIdx.x * blockDim.x + threadIdx.x;
  if (i < e) atomicAdd(&deg[dst[i]], 1.0f);
}
__global__ void k_dinv(float* __restrict__ deg, int n) {
  int i = blockIdx.x * blockDim.x + threadIdx.x;
  if (i < n) deg[i] = rsqrtf(deg[i]);
}

// ---------- propagation (F = 64) ----------
__global__ void k_selfloop64(const float* __restrict__ x, const float* __restrict__ dinv,
                             float* __restrict__ out, int total /* = N*64 */) {
  int i = blockIdx.x * blockDim.x + threadIdx.x;
  if (i >= total) return;
  int node = i >> 6;
  float di = dinv[node];
  out[i] = x[i] * di * di;
}
__global__ void k_edge_scatter64(const int* __restrict__ src, const int* __restrict__ dst,
                                 const float* __restrict__ dinv, const float* __restrict__ x,
                                 float* __restrict__ out, long total /* = E*64 */) {
  long i = (long)blockIdx.x * blockDim.x + threadIdx.x;
  if (i >= total) return;
  int e = (int)(i >> 6);
  int f = (int)(i & 63);
  int s = src[e], d = dst[e];
  float nrm = dinv[s] * dinv[d];
  atomicAdd(&out[(long)d * 64 + f], x[(long)s * 64 + f] * nrm);
}
__global__ void k_bias_relu64(float* __restrict__ y, const float* __restrict__ b, int total) {
  int i = blockIdx.x * blockDim.x + threadIdx.x;
  if (i >= total) return;
  float v = y[i] + b[i & 63];
  y[i] = v > 0.0f ? v : 0.0f;
}

// ---------- dense GEMM: Y[N x H] = X[N x K] @ Wt^T, bf16 WMMA / f32 acc ----------
template <int K, int H>
__global__ __launch_bounds__(H * 2) void k_gemm_wmma(const float* __restrict__ X,
                                                     const __bf16* __restrict__ Wt,
                                                     float* __restrict__ Y) {
  const int lane = threadIdx.x & 31;
  const int wave = threadIdx.x >> 5;          // column tile
  const int half = lane >> 4;
  const int lm   = lane & 15;
  const long rbase = (long)blockIdx.x * 16;
  const float*  __restrict__ xrow = X  + (rbase + lm) * K;          // A: lane = M
  const __bf16* __restrict__ wrow = Wt + (long)(wave * 16 + lm) * K; // B: lane = N
  v8f acc = {};
#pragma unroll
  for (int k0 = 0; k0 < K; k0 += 32) {
    const int ka = k0 + half * 8;
    acc = __builtin_amdgcn_wmma_f32_16x16x32_bf16(false, load_a_f32(xrow + ka),
                                                  false, load_b_bf16(wrow + ka),
                                                  (short)0, acc, false, false);
  }
  float* __restrict__ yb = Y + rbase * H + wave * 16 + lm;
#pragma unroll
  for (int r = 0; r < 8; ++r)                  // D: VGPR r -> M = r + half*8
    yb[(long)(r + half * 8) * H] = acc[r];
}

// ---------- head: out = [H1|H2] @ Wlin + blin, Wlt is bf16 [48 x 128] ----------
__global__ __launch_bounds__(96) void k_gemm_head(const float* __restrict__ H1,
                                                  const float* __restrict__ H2,
                                                  const __bf16* __restrict__ Wlt,
                                                  const float* __restrict__ bpad,
                                                  float* __restrict__ out, int nNodes) {
  const int lane = threadIdx.x & 31;
  const int wave = threadIdx.x >> 5;          // 0..2 -> cols 0,16,32
  const int half = lane >> 4;
  const int lm   = lane & 15;
  const long rbase = (long)blockIdx.x * 16;
  const float*  __restrict__ r1   = H1 + (rbase + lm) * 64;
  const float*  __restrict__ r2   = H2 + (rbase + lm) * 64;
  const __bf16* __restrict__ wrow = Wlt + (long)(wave * 16 + lm) * 128;
  v8f acc = {};
#pragma unroll
  for (int k0 = 0; k0 < 64; k0 += 32) {        // first 64 rows of Wlin <- H1
    const int ka = k0 + half * 8;
    acc = __builtin_amdgcn_wmma_f32_16x16x32_bf16(false, load_a_f32(r1 + ka),
                                                  false, load_b_bf16(wrow + ka),
                                                  (short)0, acc, false, false);
  }
#pragma unroll
  for (int k0 = 64; k0 < 128; k0 += 32) {      // last 64 rows of Wlin <- H2
    const int ka = k0 + half * 8;
    acc = __builtin_amdgcn_wmma_f32_16x16x32_bf16(false, load_a_f32(r2 + (ka - 64)),
                                                  false, load_b_bf16(wrow + ka),
                                                  (short)0, acc, false, false);
  }
  const int col = wave * 16 + lm;
  const float bb = bpad[col];
  if (col < 40) {
#pragma unroll
    for (int r = 0; r < 8; ++r) {
      long orow = rbase + r + half * 8;
      float v = acc[r] + bb;
      out[orow * 40 + col] = v;                       // tuple element 0
      out[(long)nNodes * 40 + orow * 40 + col] = v;   // tuple element 1
    }
  }
}

// ---------- driver ----------
extern "C" void kernel_launch(void* const* d_in, const int* in_sizes, int n_in,
                              void* d_out, int out_size, void* d_ws, size_t ws_size,
                              hipStream_t stream) {
  (void)n_in; (void)out_size; (void)ws_size;
  const float* x    = (const float*)d_in[0];
  const int*   ei   = (const int*)d_in[1];
  const float* W1   = (const float*)d_in[2];
  const float* b1   = (const float*)d_in[3];
  const float* W2   = (const float*)d_in[4];
  const float* b2   = (const float*)d_in[5];
  const float* Wlin = (const float*)d_in[6];
  const float* blin = (const float*)d_in[7];
  float* out = (float*)d_out;

  const int nN = in_sizes[0] / 128;    // 100000 (divisible by 16)
  const int nE = in_sizes[1] / 2;      // 1600000
  const int* src = ei;
  const int* dst = ei + nE;

  float* ws   = (float*)d_ws;
  float* dinv = ws;                       // [N]
  float* P0   = dinv + nN;                // [N x 64]
  float* P1   = P0 + (long)nN * 64;       // [N x 64]
  float* P2   = P1 + (long)nN * 64;       // [N x 64]
  __bf16* W1t = (__bf16*)(P2 + (long)nN * 64);  // [64 x 128] bf16
  __bf16* W2t = W1t + 64 * 128;                 // [64 x 64]  bf16
  __bf16* Wlt = W2t + 64 * 64;                  // [48 x 128] bf16 (zero-padded cols)
  float*  bpad = (float*)(Wlt + 48 * 128);      // [48] f32

  const int elemN = nN * 64;
  const long elemE = (long)nE * 64;
  auto gr = [](long n, int b) { return dim3((unsigned)((n + b - 1) / b)); };
  const dim3 B(256);
  const int rowTiles = nN / 16;

  // weight prep (bf16, transposed, padded)
  k_wt_bf16<<<gr(64 * 128, 256), B, 0, stream>>>(W1,   W1t, 128, 64, 64);
  k_wt_bf16<<<gr(64 * 64,  256), B, 0, stream>>>(W2,   W2t,  64, 64, 64);
  k_wt_bf16<<<gr(48 * 128, 256), B, 0, stream>>>(Wlin, Wlt, 128, 40, 48);
  k_pad_bias<<<1, 64, 0, stream>>>(blin, bpad, 40, 48);

  // D^{-1/2} (self-loops included)
  k_init_deg   <<<gr(nN, 256), B, 0, stream>>>(dinv, nN);
  k_deg_scatter<<<gr(nE, 256), B, 0, stream>>>(dst, dinv, nE);
  k_dinv       <<<gr(nN, 256), B, 0, stream>>>(dinv, nN);

  // layer 1: xw1 = x@W1 -> P0 ; x1 = relu(prop(P0)+b1) -> P1
  k_gemm_wmma<128, 64><<<rowTiles, 128, 0, stream>>>(x, W1t, P0);
  k_selfloop64    <<<gr(elemN, 256), B, 0, stream>>>(P0, dinv, P1, elemN);
  k_edge_scatter64<<<gr(elemE, 256), B, 0, stream>>>(src, dst, dinv, P0, P1, elemE);
  k_bias_relu64   <<<gr(elemN, 256), B, 0, stream>>>(P1, b1, elemN);

  // layer 2: xw2 = x1@W2 -> P0 ; x2 = relu(prop(P0)+b2) -> P2
  k_gemm_wmma<64, 64><<<rowTiles, 128, 0, stream>>>(P1, W2t, P0);
  k_selfloop64    <<<gr(elemN, 256), B, 0, stream>>>(P0, dinv, P2, elemN);
  k_edge_scatter64<<<gr(elemE, 256), B, 0, stream>>>(src, dst, dinv, P0, P2, elemE);
  k_bias_relu64   <<<gr(elemN, 256), B, 0, stream>>>(P2, b2, elemN);

  // JK cat + APPNP(K=1, alpha=0): propagate x1 -> P0, x2 -> P1 (column-split)
  k_selfloop64    <<<gr(elemN, 256), B, 0, stream>>>(P1, dinv, P0, elemN);
  k_edge_scatter64<<<gr(elemE, 256), B, 0, stream>>>(src, dst, dinv, P1, P0, elemE);
  k_selfloop64    <<<gr(elemN, 256), B, 0, stream>>>(P2, dinv, P1, elemN);
  k_edge_scatter64<<<gr(elemE, 256), B, 0, stream>>>(src, dst, dinv, P2, P1, elemE);

  // head: out = [P0|P1] @ Wlin + blin (written twice for the tuple)
  k_gemm_head<<<rowTiles, 96, 0, stream>>>(P0, P1, Wlt, bpad, out, nN);
}
